// EncoderWithInputAttention_47485158424751
// MI455X (gfx1250) — matrix operand order repaired and checked by reading, not compiled, not run on hardware
//
#include <hip/hip_runtime.h>
#include <hip/hip_bf16.h>

#define B_DIM  128
#define T_DIM  128
#define N_DIM  512
#define H_DIM  512
#define H2_DIM 1024
#define H4_DIM 2048

typedef __attribute__((ext_vector_type(16))) __bf16 v16bf;
typedef __attribute__((ext_vector_type(8)))  __bf16 v8bf;
typedef __attribute__((ext_vector_type(8)))  float  v8f;

// ---- WMMA operand loaders (wave32; contiguous k order, same for A and B) ----
// A: 16x32 bf16, row-major [M][K]. Lane holds row (lane&15);
//   elements 0..15 = K[(lane>>4)*16 .. +15] contiguous.
__device__ __forceinline__ v16bf load_a_bf16(const __bf16* __restrict__ row, int kk, int hlf) {
  const __bf16* p = row + kk + hlf * 16;
  v8bf lo = *(const v8bf*)(p);
  v8bf hi = *(const v8bf*)(p + 8);
  v16bf r;
#pragma unroll
  for (int i = 0; i < 8; ++i) { r[i] = lo[i]; r[i + 8] = hi[i]; }
  return r;
}
// B: 32x16 bf16 from B^T buffer [col][K]. Lane holds column (lane&15),
// K = (lane>>4)*16 .. +15 contiguous -> one 32B load.
__device__ __forceinline__ v16bf load_b_bf16(const __bf16* __restrict__ col, int kk, int hlf) {
  return *(const v16bf*)(col + kk + hlf * 16);
}

// Hardware tanh (gfx1250 V_TANH_F32, a TRANS op).
__device__ __forceinline__ float tanh_hw(float x) {
#if __has_builtin(__builtin_amdgcn_tanhf)
  return __builtin_amdgcn_tanhf(x);
#else
  float y;
  // TRANS hazard rule (ISA 7.4): pad with V_NOPs before result use.
  asm volatile("v_tanh_f32 %0, %1\n\tv_nop\n\tv_nop" : "=v"(y) : "v"(x));
  return y;
#endif
}

__device__ __forceinline__ float sigmoidf_fast(float v) {
  return 1.0f / (1.0f + __expf(-v));
}

// ---- one-time prep kernels ----
__global__ void __launch_bounds__(256) f32_to_bf16_kernel(const float* __restrict__ in,
                                                          __bf16* __restrict__ out, int n) {
  int i = blockIdx.x * 256 + threadIdx.x;
  if (i < n) out[i] = (__bf16)in[i];
}

// x [B,T,N] f32 -> xT [B,N,T] bf16 (A-matrix layout for the Ux GEMM)
__global__ void __launch_bounds__(256) transpose_x_kernel(const float* __restrict__ x,
                                                          __bf16* __restrict__ xT) {
  int idx = blockIdx.x * 256 + threadIdx.x;          // over B*T*N
  int n = idx % N_DIM;
  int bt = idx / N_DIM;
  int t = bt % T_DIM;
  int b = bt / T_DIM;
  xT[((size_t)b * N_DIM + n) * T_DIM + t] = (__bf16)x[idx];
}

__global__ void __launch_bounds__(256) init_state_kernel(float* __restrict__ c_st,
                                                         __bf16* __restrict__ hs_bf) {
  int idx = blockIdx.x * 256 + threadIdx.x;          // over B*2H
  if (idx < B_DIM * H_DIM) c_st[idx] = 0.0f;
  if (idx < B_DIM * H2_DIM) hs_bf[idx] = (__bf16)0.0f;
}

// ---- Ux[b,n,s] = sum_t x[b,t,n] * Ue_w[s,t] + Ue_b[s] ----
// grid (8, B), block (32,8). Each wave: 4 N-tiles x 1 S-tile (64x16 of [N x T]),
// sharing the B fragment across the 4 WMMAs.
__global__ void __launch_bounds__(256) ux_gemm_kernel(const __bf16* __restrict__ xT,
                                                      const __bf16* __restrict__ UeW_bf,
                                                      const float* __restrict__ Ue_b,
                                                      float* __restrict__ Ux) {
  int b      = blockIdx.y;
  int tile   = blockIdx.x * 8 + threadIdx.y;         // 0..63
  int ngroup = tile >> 3;                            // 0..7  -> rows ngroup*64
  int stile  = tile & 7;                             // 0..7
  int lane   = threadIdx.x;
  int m      = lane & 15;
  int hlf    = lane >> 4;
  const __bf16* Abase = xT + (size_t)b * N_DIM * T_DIM + (size_t)(ngroup * 64 + m) * T_DIM;
  const __bf16* Bt    = UeW_bf + (size_t)(stile * 16 + m) * T_DIM;   // Ue_w[s][t] == B^T
  v8f acc[4] = {};
#pragma unroll
  for (int kk = 0; kk < T_DIM; kk += 32) {
    v16bf bm = load_b_bf16(Bt, kk, hlf);
#pragma unroll
    for (int q = 0; q < 4; ++q) {
      v16bf a = load_a_bf16(Abase + (size_t)q * 16 * T_DIM, kk, hlf);
      acc[q] = __builtin_amdgcn_wmma_f32_16x16x32_bf16(false, a, false, bm, (short)0, acc[q],
                                                       false, false);
    }
  }
  float* outp = Ux + (size_t)b * N_DIM * T_DIM;
  int col = stile * 16 + m;
  float bias = Ue_b[col];
#pragma unroll
  for (int q = 0; q < 4; ++q)
#pragma unroll
    for (int r = 0; r < 8; ++r)
      outp[(size_t)(ngroup * 64 + q * 16 + hlf * 8 + r) * T_DIM + col] = acc[q][r] + bias;
}

// ---- w[b,s] = hs[b,:] . We_w[s,:] + We_b[s], hs = [h|c] (bf16, [B,2H]) ----
// grid: 8 blocks x 8 waves = 64 tiles of [B x T] (keep full wave parallelism,
// K=1024 is the long pole here).
__global__ void __launch_bounds__(256) attn_w_gemm_kernel(const __bf16* __restrict__ hs_bf,
                                                          const __bf16* __restrict__ WeW_bf,
                                                          const float* __restrict__ We_b,
                                                          float* __restrict__ wbuf) {
  int tile  = blockIdx.x * 8 + threadIdx.y;
  int mtile = tile >> 3;
  int stile = tile & 7;
  int lane  = threadIdx.x;
  int m     = lane & 15;
  int hlf   = lane >> 4;
  const __bf16* A  = hs_bf + (size_t)(mtile * 16 + m) * H2_DIM;
  const __bf16* Bt = WeW_bf + (size_t)(stile * 16 + m) * H2_DIM;  // We_w[s][k] == B^T
  v8f acc = {};
#pragma unroll 4
  for (int kk = 0; kk < H2_DIM; kk += 32) {
    v16bf a  = load_a_bf16(A, kk, hlf);
    v16bf bm = load_b_bf16(Bt, kk, hlf);
    acc = __builtin_amdgcn_wmma_f32_16x16x32_bf16(false, a, false, bm, (short)0, acc, false, false);
  }
  int col = stile * 16 + m;
  float bias = We_b[col];
#pragma unroll
  for (int r = 0; r < 8; ++r)
    wbuf[(size_t)(mtile * 16 + hlf * 8 + r) * T_DIM + col] = acc[r] + bias;
}

// ---- e[b,n] = sum_s tanh(w[b,s]+Ux[b,n,s])*ve_w[s] + ve_b; softmax over n;
//      xt[b,n] = alpha * x[b,t,n] (bf16). One block per b, 512 threads. ----
__global__ void __launch_bounds__(512) attn_softmax_kernel(const float* __restrict__ wbuf,
                                                           const float* __restrict__ Ux,
                                                           const float* __restrict__ ve_w,
                                                           const float* __restrict__ ve_b,
                                                           const float* __restrict__ x,
                                                           int t, __bf16* __restrict__ xt_bf) {
  __shared__ float w_s[T_DIM];
  __shared__ float ve_s[T_DIM];
  __shared__ float red[16];
  int b = blockIdx.x;
  int n = threadIdx.x;                               // 0..511
  if (n < T_DIM) { w_s[n] = wbuf[b * T_DIM + n]; ve_s[n] = ve_w[n]; }
  __syncthreads();
  const float* ux = Ux + ((size_t)b * N_DIM + n) * T_DIM;
  float acc = 0.0f;
#pragma unroll 8
  for (int s = 0; s < T_DIM; s += 4) {
    float4 u = *(const float4*)(ux + s);
    acc += tanh_hw(w_s[s + 0] + u.x) * ve_s[s + 0];
    acc += tanh_hw(w_s[s + 1] + u.y) * ve_s[s + 1];
    acc += tanh_hw(w_s[s + 2] + u.z) * ve_s[s + 2];
    acc += tanh_hw(w_s[s + 3] + u.w) * ve_s[s + 3];
  }
  float e = acc + ve_b[0];
  // block softmax over 512 lanes (16 waves of 32)
  float mv = e;
#pragma unroll
  for (int off = 16; off > 0; off >>= 1) mv = fmaxf(mv, __shfl_xor(mv, off));
  if ((n & 31) == 0) red[n >> 5] = mv;
  __syncthreads();
  float gmax = red[0];
#pragma unroll
  for (int i = 1; i < 16; ++i) gmax = fmaxf(gmax, red[i]);
  float ex = __expf(e - gmax);
  __syncthreads();
  float sv = ex;
#pragma unroll
  for (int off = 16; off > 0; off >>= 1) sv += __shfl_xor(sv, off);
  if ((n & 31) == 0) red[n >> 5] = sv;
  __syncthreads();
  float gsum = 0.0f;
#pragma unroll
  for (int i = 0; i < 16; ++i) gsum += red[i];
  float alpha = ex / gsum;
  float xv = x[((size_t)b * T_DIM + t) * N_DIM + n];
  xt_bf[(size_t)b * N_DIM + n] = (__bf16)(alpha * xv);
}

// ---- gates[b,j] = xt.W_ih^T + h.W_hh^T + b_ih + b_hh ; [B x 4H] ----
// Each wave: 1 M-tile x 4 J-tiles (16x64), A fragment reused across 4 WMMAs.
// 256 wave-tiles -> grid 32 blocks x 8 waves.
__global__ void __launch_bounds__(256) gates_gemm_kernel(const __bf16* __restrict__ xt_bf,
                                                         const __bf16* __restrict__ hs_bf,
                                                         const __bf16* __restrict__ Wih_bf,
                                                         const __bf16* __restrict__ Whh_bf,
                                                         const float* __restrict__ b_ih,
                                                         const float* __restrict__ b_hh,
                                                         float* __restrict__ gates) {
  int g      = blockIdx.x * 8 + threadIdx.y;         // 0..255
  int mtile  = g >> 5;                               // 0..7
  int jgroup = g & 31;                               // 0..31 -> cols jgroup*64
  int lane   = threadIdx.x;
  int m      = lane & 15;
  int hlf    = lane >> 4;
  v8f acc[4] = {};
  { // xt @ W_ih^T : K = N
    const __bf16* A  = xt_bf + (size_t)(mtile * 16 + m) * N_DIM;
    const __bf16* Bt = Wih_bf + (size_t)(jgroup * 64 + m) * N_DIM;  // W_ih[j][n] == B^T
#pragma unroll 2
    for (int kk = 0; kk < N_DIM; kk += 32) {
      v16bf a = load_a_bf16(A, kk, hlf);
#pragma unroll
      for (int q = 0; q < 4; ++q) {
        v16bf bm = load_b_bf16(Bt + (size_t)q * 16 * N_DIM, kk, hlf);
        acc[q] = __builtin_amdgcn_wmma_f32_16x16x32_bf16(false, a, false, bm, (short)0, acc[q],
                                                         false, false);
      }
    }
  }
  { // h @ W_hh^T : K = H (h = first half of hs, stride 2H)
    const __bf16* A  = hs_bf + (size_t)(mtile * 16 + m) * H2_DIM;
    const __bf16* Bt = Whh_bf + (size_t)(jgroup * 64 + m) * H_DIM;  // W_hh[j][h] == B^T
#pragma unroll 2
    for (int kk = 0; kk < H_DIM; kk += 32) {
      v16bf a = load_a_bf16(A, kk, hlf);
#pragma unroll
      for (int q = 0; q < 4; ++q) {
        v16bf bm = load_b_bf16(Bt + (size_t)q * 16 * H_DIM, kk, hlf);
        acc[q] = __builtin_amdgcn_wmma_f32_16x16x32_bf16(false, a, false, bm, (short)0, acc[q],
                                                         false, false);
      }
    }
  }
#pragma unroll
  for (int q = 0; q < 4; ++q) {
    int col = jgroup * 64 + q * 16 + m;
    float bias = b_ih[col] + b_hh[col];
#pragma unroll
    for (int r = 0; r < 8; ++r)
      gates[(size_t)(mtile * 16 + hlf * 8 + r) * H4_DIM + col] = acc[q][r] + bias;
  }
}

// ---- LSTM cell elementwise; writes h to output and refreshes hs_bf/c ----
__global__ void __launch_bounds__(256) lstm_cell_kernel(const float* __restrict__ gates,
                                                        float* __restrict__ c_st,
                                                        __bf16* __restrict__ hs_bf,
                                                        float* __restrict__ out, int t) {
  int idx = blockIdx.x * 256 + threadIdx.x;          // over B*H
  int j = idx % H_DIM;
  int b = idx / H_DIM;
  const float* g = gates + (size_t)b * H4_DIM;
  float gi = g[j];
  float gf = g[H_DIM + j];
  float gg = g[2 * H_DIM + j];
  float go = g[3 * H_DIM + j];
  float c_old = c_st[idx];
  float c_new = sigmoidf_fast(gf) * c_old + sigmoidf_fast(gi) * tanh_hw(gg);
  float h_new = sigmoidf_fast(go) * tanh_hw(c_new);
  c_st[idx] = c_new;
  hs_bf[(size_t)b * H2_DIM + j]         = (__bf16)h_new;
  hs_bf[(size_t)b * H2_DIM + H_DIM + j] = (__bf16)c_new;
  out[((size_t)b * T_DIM + t) * H_DIM + j] = h_new;
}

extern "C" void kernel_launch(void* const* d_in, const int* in_sizes, int n_in,
                              void* d_out, int out_size, void* d_ws, size_t ws_size,
                              hipStream_t stream) {
  const float* x    = (const float*)d_in[0];
  const float* W_ih = (const float*)d_in[1];
  const float* W_hh = (const float*)d_in[2];
  const float* b_ih = (const float*)d_in[3];
  const float* b_hh = (const float*)d_in[4];
  const float* Ue_w = (const float*)d_in[5];
  const float* Ue_b = (const float*)d_in[6];
  const float* We_w = (const float*)d_in[7];
  const float* We_b = (const float*)d_in[8];
  const float* ve_w = (const float*)d_in[9];
  const float* ve_b = (const float*)d_in[10];
  float* out = (float*)d_out;
  (void)in_sizes; (void)n_in; (void)out_size; (void)ws_size;

  char* ws = (char*)d_ws;
  size_t off = 0;
  auto carve = [&](size_t bytes) -> char* {
    char* p = ws + off;
    off += (bytes + 255) & ~(size_t)255;
    return p;
  };
  float*  Ux     = (float*) carve((size_t)B_DIM * N_DIM * T_DIM * 4);   // 33.5 MB
  __bf16* xT_bf  = (__bf16*)carve((size_t)B_DIM * N_DIM * T_DIM * 2);   // 16.8 MB
  __bf16* Wih_bf = (__bf16*)carve((size_t)H4_DIM * N_DIM * 2);          //  2.0 MB
  __bf16* Whh_bf = (__bf16*)carve((size_t)H4_DIM * H_DIM * 2);          //  2.0 MB
  __bf16* WeW_bf = (__bf16*)carve((size_t)T_DIM * H2_DIM * 2);          //  0.25 MB
  __bf16* UeW_bf = (__bf16*)carve((size_t)T_DIM * T_DIM * 2);           //  32 KB
  __bf16* hs_bf  = (__bf16*)carve((size_t)B_DIM * H2_DIM * 2);          //  [h|c] bf16
  __bf16* xt_bf  = (__bf16*)carve((size_t)B_DIM * N_DIM * 2);
  float*  c_st   = (float*) carve((size_t)B_DIM * H_DIM * 4);
  float*  wbuf   = (float*) carve((size_t)B_DIM * T_DIM * 4);
  float*  gates  = (float*) carve((size_t)B_DIM * H4_DIM * 4);

  // one-time: weight casts, x transpose, Ux GEMM, state init
  f32_to_bf16_kernel<<<(H4_DIM * N_DIM) / 256, 256, 0, stream>>>(W_ih, Wih_bf, H4_DIM * N_DIM);
  f32_to_bf16_kernel<<<(H4_DIM * H_DIM) / 256, 256, 0, stream>>>(W_hh, Whh_bf, H4_DIM * H_DIM);
  f32_to_bf16_kernel<<<(T_DIM * H2_DIM) / 256, 256, 0, stream>>>(We_w, WeW_bf, T_DIM * H2_DIM);
  f32_to_bf16_kernel<<<(T_DIM * T_DIM) / 256, 256, 0, stream>>>(Ue_w, UeW_bf, T_DIM * T_DIM);
  transpose_x_kernel<<<(B_DIM * T_DIM * N_DIM) / 256, 256, 0, stream>>>(x, xT_bf);
  ux_gemm_kernel<<<dim3(8, B_DIM), dim3(32, 8), 0, stream>>>(xT_bf, UeW_bf, Ue_b, Ux);
  init_state_kernel<<<(B_DIM * H2_DIM) / 256, 256, 0, stream>>>(c_st, hs_bf);

  // recurrent scan
  for (int t = 0; t < T_DIM; ++t) {
    attn_w_gemm_kernel<<<dim3(8), dim3(32, 8), 0, stream>>>(hs_bf, WeW_bf, We_b, wbuf);
    attn_softmax_kernel<<<dim3(B_DIM), dim3(512), 0, stream>>>(wbuf, Ux, ve_w, ve_b, x, t, xt_bf);
    gates_gemm_kernel<<<dim3(32), dim3(32, 8), 0, stream>>>(xt_bf, hs_bf, Wih_bf, Whh_bf,
                                                            b_ih, b_hh, gates);
    lstm_cell_kernel<<<(B_DIM * H_DIM) / 256, 256, 0, stream>>>(gates, c_st, hs_bf, out, t);
  }
}
